// Didi_44676249813279
// MI455X (gfx1250) — compile-verified, matrix operand order checked by reading, not run on hardware
//
#include <hip/hip_runtime.h>

// ---------------- constants ----------------
#define BATCH   32768
#define HID     512
#define DX      16
#define NSTEP   128
#define ROWS    64          // batch rows per block
#define STRIDE  520         // halves per row in activation LDS buffers (padded)
#define IN_STRIDE 80        // halves per row in input LDS buffer (cols 0..63 used)

typedef _Float16 h16 __attribute__((ext_vector_type(16)));
typedef float    f8  __attribute__((ext_vector_type(8)));
typedef unsigned int u32;
typedef u32 u32x4 __attribute__((ext_vector_type(4)));

// LDS (addrspace 3) pointer types: 32-bit addressing, DS-immediate-friendly
typedef __attribute__((address_space(3))) _Float16   lds_half;
typedef __attribute__((address_space(3))) const u32x4 lds_cvec;

union Frag { h16 v; u32x4 u[2]; };

__device__ __forceinline__ lds_half* lds_cast(_Float16* p) {
    // generic->LDS: low 32 bits of the generic address are the LDS byte address
    return (lds_half*)(u32)(unsigned long long)p;
}

// ---------------- WMMA wrapper ----------------
__device__ __forceinline__ f8 wmma16x16x32(const Frag& a, const Frag& b, f8 c) {
    return __builtin_amdgcn_wmma_f32_16x16x32_f16(
        false, a.v, false, b.v, (short)0, c, false, false);
}

// ---------------- deterministic gaussian noise ----------------
__device__ __forceinline__ u32 hashu(u32 x) {
    x ^= x >> 16; x *= 0x7feb352dU;
    x ^= x >> 15; x *= 0x846ca68bU;
    x ^= x >> 16; return x;
}
__device__ __forceinline__ float gaussian(int step, int row, int col) {
    u32 s1 = hashu((u32)step * 0x9E3779B9u ^ hashu((u32)row * 0x85EBCA6Bu + (u32)col + 0x42u));
    u32 s2 = hashu(s1 ^ 0x68E31DA4u);
    float u1 = ((float)(s1 >> 8) + 1.0f) * (1.0f / 16777216.0f);   // (0,1]
    float u2 = (float)(s2 >> 8) * (1.0f / 16777216.0f);            // [0,1)
    float r  = sqrtf(-2.0f * __logf(u1));
    return r * __cosf(6.2831853071795864f * u2);
}

// ---------------- fused dense layer: (64 x K) @ (K x 512) + b, optional relu ----------------
// inBase  = lds activation buffer + (lane&15)*INSTRIDE + (lane>>4)*8    (A-frag lane base)
// outBase = lds output buffer + (lane>>4)*8*STRIDE + ntg*16 + (lane&15) (D store lane base)
// wpB     = packed weights + (ntg<<9) + lane*16                         (B-frag lane base)
// biasB   = bias + ntg*16 + (lane&15)
// All inner-loop addresses are base + compile-time constant -> immediate offsets.
template<int KT, int INSTRIDE, bool RELU>
__device__ void layer512(const lds_half* inBase, lds_half* outBase,
                         const _Float16* __restrict__ wpB,
                         const float* __restrict__ biasB) {
    f8 acc[16];
    #pragma unroll
    for (int nt = 0; nt < 4; nt++) {
        const float bv = biasB[nt * 16];
        #pragma unroll
        for (int m = 0; m < 4; m++)
            #pragma unroll
            for (int r = 0; r < 8; r++) acc[m * 4 + nt][r] = bv;
    }

    const _Float16* wp_it = wpB;
    const lds_half* in_it = inBase;
    #pragma unroll 2
    for (int kt = 0; kt < KT; kt++) {
        Frag b[4];
        #pragma unroll
        for (int nt = 0; nt < 4; nt++) {
            const _Float16* p = wp_it + (nt << 9);
            b[nt].u[0] = *(const u32x4*)(p);
            b[nt].u[1] = *(const u32x4*)(p + 8);
        }
        #pragma unroll
        for (int m = 0; m < 4; m++) {
            Frag a;
            const lds_half* q = in_it + m * 16 * INSTRIDE;
            a.u[0] = *(lds_cvec*)(q);
            a.u[1] = *(lds_cvec*)(q + 16);
            #pragma unroll
            for (int nt = 0; nt < 4; nt++)
                acc[m * 4 + nt] = wmma16x16x32(a, b[nt], acc[m * 4 + nt]);
        }
        wp_it += (32 << 9);   // next kt row of tiles
        in_it += 32;          // next K chunk
    }

    #pragma unroll
    for (int m = 0; m < 4; m++)
        #pragma unroll
        for (int nt = 0; nt < 4; nt++)
            #pragma unroll
            for (int r = 0; r < 8; r++) {
                float x = acc[m * 4 + nt][r];
                if (RELU) x = fmaxf(x, 0.0f);      // single v_max_num_f32
                outBase[(m * 16 + r) * STRIDE + nt * 16] = (_Float16)x;
            }
}

// ---------------- main persistent-step kernel ----------------
__global__ __launch_bounds__(256) void didi_flow_kernel(
    const float* __restrict__ x1, const _Float16* __restrict__ wp_all,
    const float* __restrict__ b0, const float* __restrict__ b1,
    const float* __restrict__ b2, const float* __restrict__ b3,
    const float* __restrict__ b4, float* __restrict__ out) {

    __shared__ _Float16 bufA[ROWS * STRIDE];
    __shared__ _Float16 bufB[ROWS * STRIDE];
    __shared__ _Float16 bufIn[ROWS * IN_STRIDE];
    __shared__ float    xt[ROWS * DX];

    const int tid  = threadIdx.x;
    const int wave = tid >> 5;
    const int lane = tid & 31;
    const int rowBase = blockIdx.x * ROWS;

    const _Float16* w0p = wp_all;                       //  2*32 tiles (K padded 33->64)
    const _Float16* w1p = w0p + 2  * 32 * 512;          // 16*32 tiles
    const _Float16* w2p = w1p + 16 * 32 * 512;
    const _Float16* w3p = w2p + 16 * 32 * 512;
    const _Float16* w4p = w3p + 16 * 32 * 512;          // 16*1 tiles

    lds_half* A = lds_cast(bufA);
    lds_half* B = lds_cast(bufB);
    lds_half* I = lds_cast(bufIn);

    const int laneRow = lane & 15;
    const int grp     = lane >> 4;
    const int ntg     = wave * 4;
    const int abaseS  = laneRow * STRIDE    + grp * 8;   // A-frag base, STRIDE bufs
    const int abaseI  = laneRow * IN_STRIDE + grp * 8;   // A-frag base, input buf
    const int sbase   = grp * 8 * STRIDE + ntg * 16 + laneRow;  // D store base

    const _Float16* w0B = w0p + (ntg << 9) + lane * 16;
    const _Float16* w1B = w1p + (ntg << 9) + lane * 16;
    const _Float16* w2B = w2p + (ntg << 9) + lane * 16;
    const _Float16* w3B = w3p + (ntg << 9) + lane * 16;
    const float* b0B = b0 + ntg * 16 + laneRow;
    const float* b1B = b1 + ntg * 16 + laneRow;
    const float* b2B = b2 + ntg * 16 + laneRow;
    const float* b3B = b3 + ntg * 16 + laneRow;

    // init: x_t = x_1 ; zero input buffer; fill static x_1 columns (17..32) once
    for (int i = tid; i < ROWS * IN_STRIDE; i += 256) bufIn[i] = (_Float16)0.0f;
    for (int i = tid; i < ROWS * DX; i += 256) {
        const float v = x1[rowBase * DX + i];
        xt[i] = v;
        bufIn[(i >> 4) * IN_STRIDE + 17 + (i & 15)] = (_Float16)v;
    }
    __syncthreads();

    for (int step = 0; step < NSTEP - 1; step++) {
        const float t     = (float)(127 - step) * (1.0f / 127.0f);
        const float tprev = (float)(126 - step) * (1.0f / 127.0f);
        const float dt    = t - tprev;

        // build net input: [t, x_t, x_1, 0-pad] (cols 17..32 already hold x_1)
        for (int i = tid; i < ROWS * DX; i += 256)
            bufIn[(i >> 4) * IN_STRIDE + 1 + (i & 15)] = (_Float16)xt[i];
        if (tid < ROWS) bufIn[tid * IN_STRIDE] = (_Float16)t;
        __syncthreads();

        layer512<2,  IN_STRIDE, true>(I + abaseI, B + sbase, w0B, b0B);
        __syncthreads();
        layer512<16, STRIDE,    true>(B + abaseS, A + sbase, w1B, b1B);
        __syncthreads();
        layer512<16, STRIDE,    true>(A + abaseS, B + sbase, w2B, b2B);
        __syncthreads();
        layer512<16, STRIDE,    true>(B + abaseS, A + sbase, w3B, b3B);
        __syncthreads();

        // final layer (512 -> 16) + Euler-Maruyama update, waves 0..3 (one M-tile each)
        if (wave < 4) {
            f8 acc;
            const float bv = b4[laneRow];
            #pragma unroll
            for (int r = 0; r < 8; r++) acc[r] = bv;

            const _Float16* w4B  = w4p + lane * 16;          // tiles [kt][1]
            const lds_half* aB   = A + abaseS + wave * 16 * STRIDE;
            #pragma unroll 2
            for (int kt = 0; kt < 16; kt++) {
                Frag b, a;
                const _Float16* p = w4B + (kt << 9);
                b.u[0] = *(const u32x4*)(p);
                b.u[1] = *(const u32x4*)(p + 8);
                const lds_half* q = aB + kt * 32;
                a.u[0] = *(lds_cvec*)(q);
                a.u[1] = *(lds_cvec*)(q + 16);
                acc = wmma16x16x32(a, b, acc);
            }
            const float sig = sqrtf(0.01f * tprev * dt / t);
            #pragma unroll
            for (int r = 0; r < 8; r++) {
                const int row = wave * 16 + r + (grp << 3);
                const int col = laneRow;
                float xv = xt[row * DX + col] - dt * acc[r];   // x - dt*drift
                if (step < NSTEP - 2)
                    xv += sig * gaussian(step, rowBase + row, col);
                xt[row * DX + col] = xv;
            }
        }
        __syncthreads();
    }

    for (int i = tid; i < ROWS * DX; i += 256)
        out[rowBase * DX + i] = xt[i];
}

// ---------------- weight pack: fp32 row-major (K,N) -> f16 WMMA B-tiles ----------------
// Tile = 32(K) x 16(N), 512 halves, stored [kt][nt][lane][16 halves].
__global__ void pack_w_kernel(const float* __restrict__ w, _Float16* __restrict__ out,
                              int Kpad, int Korig, int N) {
    const int idx = blockIdx.x * 256 + threadIdx.x;
    const int total = (Kpad / 32) * (N / 16) * 512;
    if (idx >= total) return;
    const int tile = idx >> 9, within = idx & 511;
    const int lane = within >> 4, hi = within & 15;
    const int ntTotal = N / 16;
    const int kt = tile / ntTotal, nt = tile % ntTotal;
    const int v = hi >> 1, h2 = hi & 1, g = lane >> 4, nl = lane & 15;
    const int kl = ((v & 4) ? 16 : 0) + ((v & 3) << 1) + h2 + (g << 3);
    const int k = kt * 32 + kl, n = nt * 16 + nl;
    const float val = (k < Korig) ? w[k * N + n] : 0.0f;
    out[idx] = (_Float16)val;
}

// ---------------- launch ----------------
extern "C" void kernel_launch(void* const* d_in, const int* in_sizes, int n_in,
                              void* d_out, int out_size, void* d_ws, size_t ws_size,
                              hipStream_t stream) {
    const float* x1 = (const float*)d_in[0];
    const float* w0 = (const float*)d_in[1];  const float* b0 = (const float*)d_in[2];
    const float* w1 = (const float*)d_in[3];  const float* b1 = (const float*)d_in[4];
    const float* w2 = (const float*)d_in[5];  const float* b2 = (const float*)d_in[6];
    const float* w3 = (const float*)d_in[7];  const float* b3 = (const float*)d_in[8];
    const float* w4 = (const float*)d_in[9];  const float* b4 = (const float*)d_in[10];
    float* out = (float*)d_out;

    _Float16* wp = (_Float16*)d_ws;
    const size_t o0 = 0;
    const size_t o1 = o0 + (size_t)2  * 32 * 512;
    const size_t o2 = o1 + (size_t)16 * 32 * 512;
    const size_t o3 = o2 + (size_t)16 * 32 * 512;
    const size_t o4 = o3 + (size_t)16 * 32 * 512;

    auto blocksFor = [](int total) { return (total + 255) / 256; };
    pack_w_kernel<<<blocksFor(2  * 32 * 512), 256, 0, stream>>>(w0, wp + o0,  64,  33, 512);
    pack_w_kernel<<<blocksFor(16 * 32 * 512), 256, 0, stream>>>(w1, wp + o1, 512, 512, 512);
    pack_w_kernel<<<blocksFor(16 * 32 * 512), 256, 0, stream>>>(w2, wp + o2, 512, 512, 512);
    pack_w_kernel<<<blocksFor(16 * 32 * 512), 256, 0, stream>>>(w3, wp + o3, 512, 512, 512);
    pack_w_kernel<<<blocksFor(16 * 1  * 512), 256, 0, stream>>>(w4, wp + o4, 512, 512,  16);

    didi_flow_kernel<<<BATCH / ROWS, 256, 0, stream>>>(x1, wp, b0, b1, b2, b3, b4, out);
}
